// MGNGraphNet_65343632441458
// MI455X (gfx1250) — compile-verified, matrix operand order checked by reading, not run on hardware
//
#include <hip/hip_runtime.h>
#include <hip/hip_bf16.h>
#include <stdint.h>

// ---------------------------------------------------------------------------
// CDNA5 (gfx1250) wave32 WMMA types
// ---------------------------------------------------------------------------
typedef __attribute__((ext_vector_type(16))) _Float16 v16h;
typedef __attribute__((ext_vector_type(8)))  float    v8f;
typedef __attribute__((ext_vector_type(4)))  unsigned int v4u;

struct HalfPack32 { v4u a, b; };

static __device__ __forceinline__ v16h make_frag(v4u a, v4u b) {
  HalfPack32 t; t.a = a; t.b = b;
  return __builtin_bit_cast(v16h, t);
}

// A fragment (16x32 f16 tile, row-major in LDS, row stride = ld halfs).
// Per ISA 7.12.2: lane (l&15)=row, hi=(l>>4): K = hi*8..hi*8+7 and 16+hi*8..+7,
// both runs contiguous -> two b128 LDS loads.
static __device__ __forceinline__ v16h load_a_frag(const _Float16* tile, int ld, int kc) {
  int lane = threadIdx.x & 31;
  int row = lane & 15, hi = lane >> 4;
  const _Float16* p = tile + row * ld + kc * 32 + hi * 8;
  return make_frag(*(const v4u*)p, *(const v4u*)(p + 16));
}

// B fragment from pre-swizzled weights: 16 contiguous halfs per lane.
static __device__ __forceinline__ v16h load_b_frag(const _Float16* wswz, int kc, int nc) {
  int lane = threadIdx.x & 31;
  const _Float16* p = wswz + ((((kc * 8 + nc) * 32) + lane) << 4);
  return make_frag(*(const v4u*)p, *(const v4u*)(p + 8));
}

// One wave computes D[16][128] += A[16][KT*32] * W[KT*32][128].
template <int KT>
static __device__ __forceinline__ void gemm_tile(v8f* acc, const _Float16* aTile, int ld,
                                                 const _Float16* wswz) {
  for (int kc = 0; kc < KT; ++kc) {
    v16h a = load_a_frag(aTile, ld, kc);
#pragma unroll
    for (int nc = 0; nc < 8; ++nc) {
      v16h b = load_b_frag(wswz, kc, nc);
      acc[nc] = __builtin_amdgcn_wmma_f32_16x16x32_f16(
          false, a, false, b, (short)0, acc[nc], false, false);
    }
  }
}

// C/D layout: element r of acc[nc] = D[r + 8*hi][nc*16 + (lane&15)].
static __device__ __forceinline__ void epilogue_store_act(const v8f* acc, const float* bias,
                                                          bool relu, _Float16* tile, int ld) {
  int lane = threadIdx.x & 31, col = lane & 15, hi = lane >> 4;
#pragma unroll
  for (int nc = 0; nc < 8; ++nc) {
    float bv = bias[nc * 16 + col];
#pragma unroll
    for (int r = 0; r < 8; ++r) {
      float v = acc[nc][r] + bv;
      if (relu) v = fmaxf(v, 0.0f);
      tile[(r + 8 * hi) * ld + nc * 16 + col] = (_Float16)v;
    }
  }
}

// Fused bias + per-row LayerNorm; row m lives in one 16-lane half at element r.
static __device__ __forceinline__ void epilogue_rowln(v8f* acc, const float* bias,
                                                      const float* lnw, const float* lnb) {
  int lane = threadIdx.x & 31, col = lane & 15;
#pragma unroll
  for (int nc = 0; nc < 8; ++nc) {
    float bv = bias[nc * 16 + col];
#pragma unroll
    for (int r = 0; r < 8; ++r) acc[nc][r] += bv;
  }
#pragma unroll
  for (int r = 0; r < 8; ++r) {
    float s = 0.f, s2 = 0.f;
#pragma unroll
    for (int nc = 0; nc < 8; ++nc) { float v = acc[nc][r]; s += v; s2 += v * v; }
#pragma unroll
    for (int m = 1; m < 16; m <<= 1) { s += __shfl_xor(s, m, 32); s2 += __shfl_xor(s2, m, 32); }
    float mean = s * (1.f / 128.f);
    float var  = fmaxf(s2 * (1.f / 128.f) - mean * mean, 0.f);
    float rstd = rsqrtf(var + 1e-5f);
#pragma unroll
    for (int nc = 0; nc < 8; ++nc) acc[nc][r] = (acc[nc][r] - mean) * rstd;
  }
#pragma unroll
  for (int nc = 0; nc < 8; ++nc) {
    float w = lnw[nc * 16 + col], b = lnb[nc * 16 + col];
#pragma unroll
    for (int r = 0; r < 8; ++r) acc[nc][r] = acc[nc][r] * w + b;
  }
}

// ---------------------------------------------------------------------------
// Weight pre-swizzle: fp32 [K][128] -> f16 B-fragment order, zero-padded to KT*32.
// out[((kc*8+nc)*32+lane)*16 + j] = W[kc*32 + Kmap(lane,j)][nc*16 + (lane&15)]
// ---------------------------------------------------------------------------
__global__ void k_swz(const float* __restrict__ W, _Float16* __restrict__ out, int K, int KT) {
  int tid = blockIdx.x * blockDim.x + threadIdx.x;
  int total = KT * 4096;
  if (tid >= total) return;
  int j = tid & 15;
  int lane = (tid >> 4) & 31;
  int nc = (tid >> 9) & 7;
  int kc = tid >> 12;
  int hi = lane >> 4;
  int col = (lane & 15) + nc * 16;
  int kl = (j < 8) ? (hi * 8 + j) : (16 + hi * 8 + (j - 8));
  int k = kc * 32 + kl;
  float v = (k < K) ? W[(size_t)k * 128 + col] : 0.f;
  out[tid] = (_Float16)v;
}

__global__ void k_edge_count(const int* __restrict__ ei, int E, float* __restrict__ cnt) {
  int i = blockIdx.x * blockDim.x + threadIdx.x;
  int stride = gridDim.x * blockDim.x;
  for (; i < E; i += stride) atomicAdd(&cnt[ei[E + i]], 1.0f);
}

// ---------------------------------------------------------------------------
// Encoder: 4-linear MLP (in -> 128 x4, ReLU after first 3), writes pre-LN output
// and accumulates global sum/sumsq for the graph LayerNorm.
// ---------------------------------------------------------------------------
__global__ __launch_bounds__(128) void k_encoder(
    const float* __restrict__ x, int nrows, int nin,
    float* __restrict__ out, float* __restrict__ stats,
    const _Float16* __restrict__ w0, const _Float16* __restrict__ w1,
    const _Float16* __restrict__ w2, const _Float16* __restrict__ w3,
    const float* __restrict__ bs /* 4x128 */) {
  __shared__ __align__(16) _Float16 A0[4][16][32];
  __shared__ __align__(16) _Float16 ACT[4][16][128];
  int wave = threadIdx.x >> 5, lane = threadIdx.x & 31;
  int n0 = (blockIdx.x * 4 + wave) * 16;
  {
    int row = lane & 15, half = lane >> 4;
    int ng = n0 + row;
#pragma unroll
    for (int j = 0; j < 16; ++j) {
      int c = half * 16 + j;
      float v = 0.f;
      if (ng < nrows && c < nin) v = x[(size_t)ng * nin + c];
      A0[wave][row][c] = (_Float16)v;
    }
  }
  __syncthreads();
  __builtin_prefetch(w1, 0, 1);
  __builtin_prefetch(w2, 0, 1);

  _Float16* tile = &ACT[wave][0][0];
  v8f a1[8] = {};
  gemm_tile<1>(a1, &A0[wave][0][0], 32, w0);
  epilogue_store_act(a1, bs, true, tile, 128);
  v8f a2[8] = {};
  gemm_tile<4>(a2, tile, 128, w1);
  epilogue_store_act(a2, bs + 128, true, tile, 128);
  v8f a3[8] = {};
  gemm_tile<4>(a3, tile, 128, w2);
  epilogue_store_act(a3, bs + 256, true, tile, 128);
  v8f a4[8] = {};
  gemm_tile<4>(a4, tile, 128, w3);

  int col = lane & 15, hi = lane >> 4;
  float s = 0.f, s2 = 0.f;
#pragma unroll
  for (int nc = 0; nc < 8; ++nc) {
    float bv = bs[384 + nc * 16 + col];
#pragma unroll
    for (int r = 0; r < 8; ++r) {
      int m = r + 8 * hi, ng = n0 + m;
      float v = a4[nc][r] + bv;
      if (ng < nrows) {
        out[(size_t)ng * 128 + nc * 16 + col] = v;
        s += v; s2 += v * v;
      }
    }
  }
#pragma unroll
  for (int m = 1; m < 32; m <<= 1) { s += __shfl_xor(s, m, 32); s2 += __shfl_xor(s2, m, 32); }
  if (lane == 0) { atomicAdd(&stats[0], s); atomicAdd(&stats[1], s2); }
}

__global__ void k_fin_stats(float* stats, float count) {
  if (threadIdx.x == 0 && blockIdx.x == 0) {
    float mu = stats[0] / count;
    float var = fmaxf(stats[1] / count - mu * mu, 0.f);
    stats[2] = mu;
    stats[3] = 1.f / (sqrtf(var) + 1e-5f);   // reference: (x-mu)/(std+eps)
  }
}

__global__ void k_apply_gln(float* __restrict__ v, size_t n, const float* __restrict__ stats,
                            const float* __restrict__ w, const float* __restrict__ b) {
  size_t i = (size_t)blockIdx.x * blockDim.x + threadIdx.x;
  size_t stride = (size_t)gridDim.x * blockDim.x;
  float mu = stats[2], sc = stats[3];
  for (; i < n; i += stride) {
    int f = (int)(i & 127);
    v[i] = (v[i] - mu) * sc * w[f] + b[f];
  }
}

// ---------------------------------------------------------------------------
// Edge message: gather [h[dst] | h[src] | e] -> MLP3 -> rowLN -> e += tmp,
// atomic scatter-add of tmp into agg[dst].
// ---------------------------------------------------------------------------
__global__ __launch_bounds__(128) void k_edge_msg(
    const float* __restrict__ h, float* __restrict__ e, float* __restrict__ agg,
    const int* __restrict__ ei, int E,
    const _Float16* __restrict__ w0, const _Float16* __restrict__ w1,
    const _Float16* __restrict__ w2,
    const float* __restrict__ bs,      // 3*128
    const float* __restrict__ lnw, const float* __restrict__ lnb) {
  __shared__ __align__(16) _Float16 A[4][16][384];
  __shared__ int eidx[4][16][2];
  int wave = threadIdx.x >> 5, lane = threadIdx.x & 31;
  int e0 = (blockIdx.x * 4 + wave) * 16;
  if (lane < 16) {
    int eg = e0 + lane;
    int s = 0, d = 0;
    if (eg < E) { s = ei[eg]; d = ei[E + eg]; }  // edge_index[0]=src, [1]=dst
    eidx[wave][lane][0] = d;
    eidx[wave][lane][1] = s;
  }
  __syncthreads();
  for (int m = 0; m < 16; ++m) {
    int eg = e0 + m;
    bool valid = eg < E;
    int dst = eidx[wave][m][0], src = eidx[wave][m][1];
    const float* seg0 = h + (size_t)dst * 128;
    const float* seg1 = h + (size_t)src * 128;
    const float* seg2 = e + (size_t)eg * 128;
#pragma unroll
    for (int t = 0; t < 3; ++t) {
      int c = (t * 32 + lane) * 4;   // t=0: 0..127 hd, t=1: 128..255 hs, t=2: 256..383 e
      const float* sp = (t == 0) ? (seg0 + c) : (t == 1) ? (seg1 + (c - 128)) : (seg2 + (c - 256));
      float4 v = make_float4(0.f, 0.f, 0.f, 0.f);
      if (valid) v = *(const float4*)sp;
      _Float16* dp = &A[wave][m][c];
      dp[0] = (_Float16)v.x; dp[1] = (_Float16)v.y;
      dp[2] = (_Float16)v.z; dp[3] = (_Float16)v.w;
    }
  }
  __syncthreads();
  __builtin_prefetch(w1, 0, 1);
  __builtin_prefetch(w2, 0, 1);

  _Float16* tile = &A[wave][0][0];
  v8f a1[8] = {};
  gemm_tile<12>(a1, tile, 384, w0);
  epilogue_store_act(a1, bs, true, tile, 384);       // act aliases cols 0..127 (same-wave LDS is in-order)
  v8f a2[8] = {};
  gemm_tile<4>(a2, tile, 384, w1);
  epilogue_store_act(a2, bs + 128, true, tile, 384);
  v8f a3[8] = {};
  gemm_tile<4>(a3, tile, 384, w2);
  epilogue_rowln(a3, bs + 256, lnw, lnb);

  int col = lane & 15, hi = lane >> 4;
#pragma unroll
  for (int r = 0; r < 8; ++r) {
    int m = r + 8 * hi;
    int eg = e0 + m;
    if (eg < E) {
      int dst = eidx[wave][m][0];
      float* erow = e + (size_t)eg * 128;
      float* arow = agg + (size_t)dst * 128;
#pragma unroll
      for (int nc = 0; nc < 8; ++nc) {
        int f = nc * 16 + col;
        float tv = a3[nc][r];
        erow[f] += tv;                 // e = tmp + e (row owned exclusively)
        atomicAdd(&arow[f], tv);       // segment_sum(tmp, dst)
      }
    }
  }
}

// ---------------------------------------------------------------------------
// Node update: MLP3(concat(h, agg/cnt)) -> rowLN -> h += .
// ---------------------------------------------------------------------------
__global__ __launch_bounds__(128) void k_node_upd(
    float* __restrict__ h, const float* __restrict__ agg, const float* __restrict__ cnt, int N,
    const _Float16* __restrict__ w0, const _Float16* __restrict__ w1,
    const _Float16* __restrict__ w2, const float* __restrict__ bs,
    const float* __restrict__ lnw, const float* __restrict__ lnb) {
  __shared__ __align__(16) _Float16 A[4][16][256];
  int wave = threadIdx.x >> 5, lane = threadIdx.x & 31;
  int n0 = (blockIdx.x * 4 + wave) * 16;
  for (int m = 0; m < 16; ++m) {
    int ng = n0 + m;
    bool valid = ng < N;
    const float* hp = h + (size_t)ng * 128;
    const float* ap = agg + (size_t)ng * 128;
    float rc = 0.f;
    if (valid) rc = 1.f / fmaxf(cnt[ng], 1.f);
#pragma unroll
    for (int t = 0; t < 2; ++t) {
      int c = (t * 32 + lane) * 4;
      float4 v = make_float4(0.f, 0.f, 0.f, 0.f);
      if (valid) {
        if (t == 0) v = *(const float4*)(hp + c);
        else {
          v = *(const float4*)(ap + (c - 128));
          v.x *= rc; v.y *= rc; v.z *= rc; v.w *= rc;
        }
      }
      _Float16* dp = &A[wave][m][c];
      dp[0] = (_Float16)v.x; dp[1] = (_Float16)v.y;
      dp[2] = (_Float16)v.z; dp[3] = (_Float16)v.w;
    }
  }
  __syncthreads();
  _Float16* tile = &A[wave][0][0];
  v8f a1[8] = {};
  gemm_tile<8>(a1, tile, 256, w0);
  epilogue_store_act(a1, bs, true, tile, 256);
  v8f a2[8] = {};
  gemm_tile<4>(a2, tile, 256, w1);
  epilogue_store_act(a2, bs + 128, true, tile, 256);
  v8f a3[8] = {};
  gemm_tile<4>(a3, tile, 256, w2);
  epilogue_rowln(a3, bs + 256, lnw, lnb);

  int col = lane & 15, hi = lane >> 4;
#pragma unroll
  for (int r = 0; r < 8; ++r) {
    int m = r + 8 * hi, ng = n0 + m;
    if (ng < N) {
      float* hr = h + (size_t)ng * 128;
#pragma unroll
      for (int nc = 0; nc < 8; ++nc) hr[nc * 16 + col] += a3[nc][r];
    }
  }
}

// ---------------------------------------------------------------------------
// Decoder: 3x (128->128 ReLU) via WMMA, then 128->3 scalar.
// ---------------------------------------------------------------------------
__global__ __launch_bounds__(128) void k_decoder(
    const float* __restrict__ h, int N,
    const _Float16* __restrict__ w0, const _Float16* __restrict__ w1,
    const _Float16* __restrict__ w2, const float* __restrict__ bs /*3*128*/,
    const float* __restrict__ Wl /*128x3*/, const float* __restrict__ bl /*3*/,
    float* __restrict__ out) {
  __shared__ __align__(16) _Float16 ACT[4][16][128];
  int wave = threadIdx.x >> 5, lane = threadIdx.x & 31;
  int n0 = (blockIdx.x * 4 + wave) * 16;
  for (int m = 0; m < 16; ++m) {
    int ng = n0 + m;
    int c = lane * 4;
    float4 v = make_float4(0.f, 0.f, 0.f, 0.f);
    if (ng < N) v = *(const float4*)(h + (size_t)ng * 128 + c);
    _Float16* dp = &ACT[wave][m][c];
    dp[0] = (_Float16)v.x; dp[1] = (_Float16)v.y;
    dp[2] = (_Float16)v.z; dp[3] = (_Float16)v.w;
  }
  __syncthreads();
  _Float16* tile = &ACT[wave][0][0];
  v8f a1[8] = {};
  gemm_tile<4>(a1, tile, 128, w0);
  epilogue_store_act(a1, bs, true, tile, 128);
  v8f a2[8] = {};
  gemm_tile<4>(a2, tile, 128, w1);
  epilogue_store_act(a2, bs + 128, true, tile, 128);
  v8f a3[8] = {};
  gemm_tile<4>(a3, tile, 128, w2);
  epilogue_store_act(a3, bs + 256, true, tile, 128);

  if (lane < 16) {
    int ng = n0 + lane;
    if (ng < N) {
      float o0 = bl[0], o1 = bl[1], o2 = bl[2];
      for (int k = 0; k < 128; ++k) {
        float a = (float)ACT[wave][lane][k];
        o0 += a * Wl[k * 3 + 0];
        o1 += a * Wl[k * 3 + 1];
        o2 += a * Wl[k * 3 + 2];
      }
      out[(size_t)ng * 3 + 0] = o0;
      out[(size_t)ng * 3 + 1] = o1;
      out[(size_t)ng * 3 + 2] = o2;
    }
  }
}

// ---------------------------------------------------------------------------
// Host orchestration
// ---------------------------------------------------------------------------
extern "C" void kernel_launch(void* const* d_in, const int* in_sizes, int n_in,
                              void* d_out, int out_size, void* d_ws, size_t ws_size,
                              hipStream_t stream) {
  (void)n_in; (void)out_size; (void)ws_size;
  const float* x        = (const float*)d_in[0];
  const int*   ei       = (const int*)d_in[1];
  const float* ea       = (const float*)d_in[2];
  const float* encN_W0  = (const float*)d_in[3];
  const float* encN_Ws  = (const float*)d_in[4];
  const float* encN_bs  = (const float*)d_in[5];
  const float* encN_lnw = (const float*)d_in[6];
  const float* encN_lnb = (const float*)d_in[7];
  const float* encE_W0  = (const float*)d_in[8];
  const float* encE_Ws  = (const float*)d_in[9];
  const float* encE_bs  = (const float*)d_in[10];
  const float* encE_lnw = (const float*)d_in[11];
  const float* encE_lnb = (const float*)d_in[12];
  const float* eW0      = (const float*)d_in[13];
  const float* eWs      = (const float*)d_in[14];
  const float* ebs      = (const float*)d_in[15];
  const float* elnw     = (const float*)d_in[16];
  const float* elnb     = (const float*)d_in[17];
  const float* nW0      = (const float*)d_in[18];
  const float* nWs      = (const float*)d_in[19];
  const float* nbs      = (const float*)d_in[20];
  const float* nlnw     = (const float*)d_in[21];
  const float* nlnb     = (const float*)d_in[22];
  const float* dec_W0   = (const float*)d_in[23];
  const float* dec_Ws   = (const float*)d_in[24];
  const float* dec_bs   = (const float*)d_in[25];
  const float* dec_Wl   = (const float*)d_in[26];
  const float* dec_bl   = (const float*)d_in[27];

  const int N = in_sizes[0] / 5;
  const int E = in_sizes[1] / 2;
  const int L = 4;

  char* ws = (char*)d_ws;
  size_t off = 0;
  auto alloc = [&](size_t bytes) -> void* {
    off = (off + 255) & ~(size_t)255;
    void* p = ws + off;
    off += bytes;
    return p;
  };

  _Float16 *swzN[4], *swzE[4];
  swzN[0] = (_Float16*)alloc(32 * 128 * 2);
  for (int i = 1; i < 4; ++i) swzN[i] = (_Float16*)alloc(128 * 128 * 2);
  swzE[0] = (_Float16*)alloc(32 * 128 * 2);
  for (int i = 1; i < 4; ++i) swzE[i] = (_Float16*)alloc(128 * 128 * 2);
  _Float16 *swz_e0[4], *swz_e1[4], *swz_e2[4], *swz_n0[4], *swz_n1[4], *swz_n2[4];
  for (int l = 0; l < L; ++l) {
    swz_e0[l] = (_Float16*)alloc(384 * 128 * 2);
    swz_e1[l] = (_Float16*)alloc(128 * 128 * 2);
    swz_e2[l] = (_Float16*)alloc(128 * 128 * 2);
    swz_n0[l] = (_Float16*)alloc(256 * 128 * 2);
    swz_n1[l] = (_Float16*)alloc(128 * 128 * 2);
    swz_n2[l] = (_Float16*)alloc(128 * 128 * 2);
  }
  _Float16* swz_d0 = (_Float16*)alloc(128 * 128 * 2);
  _Float16* swz_d1 = (_Float16*)alloc(128 * 128 * 2);
  _Float16* swz_d2 = (_Float16*)alloc(128 * 128 * 2);

  float* h     = (float*)alloc((size_t)N * 128 * 4);
  float* e     = (float*)alloc((size_t)E * 128 * 4);
  float* agg   = (float*)alloc((size_t)N * 128 * 4);
  float* cnt   = (float*)alloc((size_t)N * 4);
  float* stats = (float*)alloc(8 * 4);

  auto swz = [&](const float* W, _Float16* dst, int K, int KT) {
    int total = KT * 4096;
    k_swz<<<(total + 255) / 256, 256, 0, stream>>>(W, dst, K, KT);
  };
  swz(encN_W0, swzN[0], 5, 1);
  for (int i = 0; i < 3; ++i) swz(encN_Ws + (size_t)i * 128 * 128, swzN[1 + i], 128, 4);
  swz(encE_W0, swzE[0], 3, 1);
  for (int i = 0; i < 3; ++i) swz(encE_Ws + (size_t)i * 128 * 128, swzE[1 + i], 128, 4);
  for (int l = 0; l < L; ++l) {
    swz(eW0 + (size_t)l * 384 * 128, swz_e0[l], 384, 12);
    swz(eWs + ((size_t)l * 2 + 0) * 128 * 128, swz_e1[l], 128, 4);
    swz(eWs + ((size_t)l * 2 + 1) * 128 * 128, swz_e2[l], 128, 4);
    swz(nW0 + (size_t)l * 256 * 128, swz_n0[l], 256, 8);
    swz(nWs + ((size_t)l * 2 + 0) * 128 * 128, swz_n1[l], 128, 4);
    swz(nWs + ((size_t)l * 2 + 1) * 128 * 128, swz_n2[l], 128, 4);
  }
  swz(dec_W0, swz_d0, 128, 4);
  swz(dec_Ws, swz_d1, 128, 4);
  swz(dec_Ws + 128 * 128, swz_d2, 128, 4);

  hipMemsetAsync(stats, 0, 8 * sizeof(float), stream);
  hipMemsetAsync(cnt, 0, (size_t)N * sizeof(float), stream);
  k_edge_count<<<256, 256, 0, stream>>>(ei, E, cnt);

  int nb_n = (N + 63) / 64;
  int nb_e = (E + 63) / 64;

  k_encoder<<<nb_n, 128, 0, stream>>>(x, N, 5, h, stats, swzN[0], swzN[1], swzN[2], swzN[3], encN_bs);
  k_encoder<<<nb_e, 128, 0, stream>>>(ea, E, 3, e, stats + 4, swzE[0], swzE[1], swzE[2], swzE[3], encE_bs);
  k_fin_stats<<<1, 1, 0, stream>>>(stats, (float)N * 128.f);
  k_fin_stats<<<1, 1, 0, stream>>>(stats + 4, (float)E * 128.f);
  k_apply_gln<<<2048, 256, 0, stream>>>(h, (size_t)N * 128, stats, encN_lnw, encN_lnb);
  k_apply_gln<<<4096, 256, 0, stream>>>(e, (size_t)E * 128, stats + 4, encE_lnw, encE_lnb);

  for (int l = 0; l < L; ++l) {
    hipMemsetAsync(agg, 0, (size_t)N * 128 * sizeof(float), stream);
    k_edge_msg<<<nb_e, 128, 0, stream>>>(h, e, agg, ei, E,
        swz_e0[l], swz_e1[l], swz_e2[l],
        ebs + (size_t)l * 3 * 128, elnw + l * 128, elnb + l * 128);
    k_node_upd<<<nb_n, 128, 0, stream>>>(h, agg, cnt, N,
        swz_n0[l], swz_n1[l], swz_n2[l],
        nbs + (size_t)l * 3 * 128, nlnw + l * 128, nlnb + l * 128);
  }

  k_decoder<<<nb_n, 128, 0, stream>>>(h, N, swz_d0, swz_d1, swz_d2, dec_bs, dec_Wl, dec_bl,
                                      (float*)d_out);
}
